// ScaledDotProductAttention_3075196584285
// MI455X (gfx1250) — matrix operand order repaired and checked by reading, not compiled, not run on hardware
//
#include <hip/hip_runtime.h>

typedef __attribute__((ext_vector_type(16))) __bf16 v16bf;
typedef __attribute__((ext_vector_type(8)))  float  v8f;
typedef unsigned int v4u __attribute__((ext_vector_type(4)));
typedef int          v4i __attribute__((ext_vector_type(4)));
typedef int          v8i __attribute__((ext_vector_type(8)));

#define SEQ   1024
#define DH    128
#define NH    8
#define NB    4
#define ROWS  32                     // query rows per block (2 WMMA M-tiles)
#define NEGINF (-1.0e12f)

__global__ __launch_bounds__(256)
void entmax15_attn_kernel(const float* __restrict__ qg,
                          const float* __restrict__ kg,
                          const float* __restrict__ vg,
                          const int*   __restrict__ maskg,
                          float* __restrict__ outg)
{
    // LDS: 32x1024 fp32 scores (128KB) + 32x128 fp32 Q stage (16KB) + stats
    __shared__ float sc[ROWS * SEQ];
    __shared__ float qf32[ROWS * DH];
    __shared__ float rowmax[ROWS];
    __shared__ float rowtau[ROWS];

    const int tid  = threadIdx.x;
    const int lane = tid & 31;
    const int wave = tid >> 5;
    const int half = lane >> 4;   // 0: lanes 0-15, 1: lanes 16-31
    const int l15  = lane & 15;

    const int qTiles = SEQ / ROWS;           // 32
    const int bh = blockIdx.x / qTiles;      // fused (b*H + h)
    const int qt = blockIdx.x % qTiles;
    const int b  = bh / NH;
    const int q0 = qt * ROWS;

    const float* qp = qg + (size_t)bh * SEQ * DH;
    const float* kp = kg + (size_t)bh * SEQ * DH;
    const float* vp = vg + (size_t)bh * SEQ * DH;
    const int*   mp = maskg + ((size_t)b * SEQ + q0) * SEQ;   // mask is (B,1,S,S)
    float*       op = outg + ((size_t)bh * SEQ + q0) * DH;

    // ---------------- phase 1: Q tile -> LDS via Tensor Data Mover -----------
    // Q rows [q0, q0+32) are one flat 16KB region; a single TDM descriptor
    // (count=1, 4-byte elements, tile 4096x1) DMAs it into LDS, freeing the
    // waves' VMEM path. One wave issues it (TDM ignores EXEC), waits on
    // TENSORcnt, then the block barrier publishes the data.
#if defined(__has_builtin) && __has_builtin(__builtin_amdgcn_tensor_load_to_lds)
    if (wave == 0) {
        const unsigned long long ga =
            (unsigned long long)(const void*)(qp + (size_t)q0 * DH);
        const unsigned int ldsa = (unsigned int)(size_t)(void*)qf32; // flat->LDS offset

        v4u g0;
        g0[0] = 1u;                                   // count=1 (valid user D#)
        g0[1] = ldsa;                                 // lds_addr (bytes)
        g0[2] = (unsigned int)ga;                     // global_addr[31:0]
        g0[3] = (unsigned int)((ga >> 32) & 0x01FFFFFFu) | (2u << 30); // [56:32] | type=2

        v8i g1;
        g1[0] = (2 << 16);            // wg_mask=0 | data_size=2 (4B) | no pad/iter
        g1[1] = (int)(4096u << 16);   // atomic_barrier_addr=0 | tensor_dim0.lo16=4096
        g1[2] = (int)(1u << 16);      // tensor_dim0.hi16=0    | tensor_dim1.lo16=1
        g1[3] = (int)(4096u << 16);   // tensor_dim1.hi16=0    | tile_dim0=4096
        g1[4] = 1;                    // tile_dim1=1 | tile_dim2=0
        g1[5] = 4096;                 // tensor_dim0_stride.lo32
        g1[6] = 0;                    // stride.hi16 | tensor_dim1_stride.lo16
        g1[7] = 0;

        v4i gz = {0, 0, 0, 0};
#if __clang_major__ >= 23
        v8i gz8 = {0, 0, 0, 0, 0, 0, 0, 0};
        __builtin_amdgcn_tensor_load_to_lds(g0, g1, gz, gz, gz8, 0);
#else
        __builtin_amdgcn_tensor_load_to_lds(g0, g1, gz, gz, 0);
#endif
        __builtin_amdgcn_s_wait_tensorcnt(0);
    }
#else
    for (int i = tid; i < ROWS * DH; i += 256) {
        int r = i >> 7, d = i & (DH - 1);
        qf32[i] = qp[(size_t)(q0 + r) * DH + d];
    }
#endif
    __syncthreads();

    // A-fragments for both 16-row M-tiles of Q (16x32 bf16 each), held for all
    // key tiles.  ISA 7.12.2 16-bit A layout: lane = M in each half; element e:
    // K = 16*(e>>3) + 8*half + 2*((e>>1)&3) + (e&1)
    v16bf aQ[8];   // [mt*4 + dstep]
    #pragma unroll
    for (int e = 0; e < 16; ++e) {
        const int dk = ((e >> 3) * 16) + half * 8 + ((e >> 1) & 3) * 2 + (e & 1);
        #pragma unroll
        for (int mt = 0; mt < 2; ++mt)
            #pragma unroll
            for (int dstep = 0; dstep < 4; ++dstep)
                aQ[mt * 4 + dstep][e] =
                    (__bf16)qf32[(mt * 16 + l15) * DH + dstep * 32 + dk];
    }

    const float scale = 0.088388347648318447f;   // 1/sqrt(128)

    // ---------------- phase 2: scores = scale*(Q K^T), masked, -> LDS --------
    for (int kt = wave; kt < SEQ / 16; kt += 8) {
        const int key0 = kt * 16;
        // prefetch this wave's next key tile (lowers to global_prefetch_b8)
        if (kt + 8 < SEQ / 16)
            __builtin_prefetch(&kp[(size_t)(key0 + 128) * DH + lane * 4], 0, 1);

        // B-fragments (32x16 bf16): lane = N col (key), element e = K (d) in A's order;
        // loaded once, reused by both M-tiles.
        v16bf bK[4];
        #pragma unroll
        for (int dstep = 0; dstep < 4; ++dstep)
            #pragma unroll
            for (int e = 0; e < 16; ++e)
                bK[dstep][e] =
                    (__bf16)kp[(size_t)(key0 + l15) * DH + dstep * 32 + ((e >> 3) * 16) + half * 8 + ((e >> 1) & 3) * 2 + (e & 1)];

        #pragma unroll
        for (int mt = 0; mt < 2; ++mt) {
            v8f c = {};
            #pragma unroll
            for (int dstep = 0; dstep < 4; ++dstep)
                c = __builtin_amdgcn_wmma_f32_16x16x32_bf16(
                        false, aQ[mt * 4 + dstep], false, bK[dstep],
                        (short)0, c, false, false);
            const int key = key0 + l15;
            #pragma unroll
            for (int r = 0; r < 8; ++r) {
                const int m = mt * 16 + r + half * 8;  // C layout: VGPR r -> M=r / r+8
                float val = c[r] * scale;
                if (mp[(size_t)m * SEQ + key] == 0) val = NEGINF;
                sc[m * SEQ + key] = val;
            }
        }
    }
    __syncthreads();

    // ---------------- phase 3: row max + exact bisection for tau -------------
    // entmax1.5: x' = (s - max)/2; p_i = clip(x'_i - tau)^2, sum p = 1.
    // sum(tau) monotone decreasing, tau* in [-1, 0]; 30 bisections -> 2^-30.
    for (int rr = 0; rr < 4; ++rr) {
        const int row = wave * 4 + rr;
        float xs[32];
        float mx = -3.0e38f;
        #pragma unroll
        for (int j = 0; j < 32; ++j) {
            xs[j] = sc[row * SEQ + lane + j * 32];
            mx = fmaxf(mx, xs[j]);
        }
        #pragma unroll
        for (int off = 16; off >= 1; off >>= 1)
            mx = fmaxf(mx, __shfl_xor(mx, off, 32));
        #pragma unroll
        for (int j = 0; j < 32; ++j) xs[j] = (xs[j] - mx) * 0.5f;

        float lo = -1.0f, hi = 0.0f;
        for (int it = 0; it < 30; ++it) {
            const float tau = 0.5f * (lo + hi);
            float s = 0.0f;
            #pragma unroll
            for (int j = 0; j < 32; ++j) {
                float t = fmaxf(xs[j] - tau, 0.0f);
                s += t * t;
            }
            #pragma unroll
            for (int off = 16; off >= 1; off >>= 1)
                s += __shfl_xor(s, off, 32);
            if (s >= 1.0f) lo = tau; else hi = tau;
        }
        if (lane == 0) { rowmax[row] = mx; rowtau[row] = 0.5f * (lo + hi); }
    }
    __syncthreads();

    // ---------------- phase 4: out = P @ V (wave owns 16 output columns) -----
    const int   n0    = wave * 16;
    const float rmax0 = rowmax[l15],      tau0 = rowtau[l15];
    const float rmax1 = rowmax[16 + l15], tau1 = rowtau[16 + l15];
    v8f co0 = {}, co1 = {};
    for (int kstep = 0; kstep < SEQ / 32; ++kstep) {
        v16bf aP0, aP1, bV;
        #pragma unroll
        for (int e = 0; e < 16; ++e) {
            const int kk = kstep * 32 + ((e >> 3) * 16) + half * 8
                         + ((e >> 1) & 3) * 2 + (e & 1);
            float x0 = fmaxf((sc[l15 * SEQ + kk] - rmax0) * 0.5f - tau0, 0.0f);
            float x1 = fmaxf((sc[(16 + l15) * SEQ + kk] - rmax1) * 0.5f - tau1, 0.0f);
            aP0[e] = (__bf16)(x0 * x0);
            aP1[e] = (__bf16)(x1 * x1);
            bV[e]  = (__bf16)vp[(size_t)kk * DH + n0 + l15];
        }
        co0 = __builtin_amdgcn_wmma_f32_16x16x32_bf16(
                  false, aP0, false, bV, (short)0, co0, false, false);
        co1 = __builtin_amdgcn_wmma_f32_16x16x32_bf16(
                  false, aP1, false, bV, (short)0, co1, false, false);
    }
    #pragma unroll
    for (int r = 0; r < 8; ++r) {
        const int m = r + half * 8;
        op[(size_t)m * DH + n0 + l15]        = co0[r];
        op[(size_t)(16 + m) * DH + n0 + l15] = co1[r];
    }
}

extern "C" void kernel_launch(void* const* d_in, const int* in_sizes, int n_in,
                              void* d_out, int out_size, void* d_ws, size_t ws_size,
                              hipStream_t stream) {
    const float* q    = (const float*)d_in[0];
    const float* k    = (const float*)d_in[1];
    const float* v    = (const float*)d_in[2];
    const int*   mask = (const int*)d_in[3];
    float*       out  = (float*)d_out;
    (void)in_sizes; (void)n_in; (void)out_size; (void)d_ws; (void)ws_size;

    const int blocks = NB * NH * (SEQ / ROWS);   // 1024
    entmax15_attn_kernel<<<blocks, 256, 0, stream>>>(q, k, v, mask, out);
}
